// Attention_55044300865733
// MI455X (gfx1250) — matrix operand order repaired and checked by reading, not compiled, not run on hardware
//
#include <hip/hip_runtime.h>
#include <hip/hip_bf16.h>

typedef __attribute__((ext_vector_type(16))) _Float16 v16h;
typedef __attribute__((ext_vector_type(8)))  _Float16 v8h;
typedef __attribute__((ext_vector_type(8)))  float    v8f;

#define DIMD   1024
#define SEQ    2048
#define BATCH  4
#define NROW   (BATCH * SEQ)          // 8192 total rows
#define ATT_SCALE 0.03125f            // 1/sqrt(1024)
#define LSTR   (SEQ + 8)              // padded LDS row stride (words), keeps v8f align + spreads banks

union AFrag { v16h v; v8h h[2]; };

// ---------------------------------------------------------------------------
// fp32 -> f16 convert (vectorized by 4)
// ---------------------------------------------------------------------------
__global__ __launch_bounds__(256) void cvt_f32_f16(const float* __restrict__ in,
                                                   _Float16* __restrict__ out, int n) {
    int i = (blockIdx.x * 256 + threadIdx.x) * 4;
    if (i < n) {
        float4 v = *(const float4*)(in + i);
        out[i + 0] = (_Float16)v.x;
        out[i + 1] = (_Float16)v.y;
        out[i + 2] = (_Float16)v.z;
        out[i + 3] = (_Float16)v.w;
    }
}

// ---------------------------------------------------------------------------
// W (fp32 [D][D], for x@W) -> Wt (f16 [D][D], Wt[n][k] = W[k][n])
// makes the WMMA B-fragment (column n, 16 contiguous K) a single 32B load
// ---------------------------------------------------------------------------
__global__ __launch_bounds__(256) void cvt_w_transpose(const float* __restrict__ W,
                                                       _Float16* __restrict__ Wt) {
    int idx = blockIdx.x * 256 + threadIdx.x;     // grid covers D*D
    int n = idx >> 10;
    int k = idx & (DIMD - 1);
    Wt[idx] = (_Float16)W[k * DIMD + n];
}

// ---------------------------------------------------------------------------
// Y = x @ W + b  via v_wmma_f32_16x16x32_f16
// xh: f16 [8192][1024]  Wt: f16 [1024][1024] (pre-transposed)  bias: f32 [1024]
// out: f16; transposeOut=0 -> [8192][1024] row-major (Q, K)
//           transposeOut=1 -> per-batch [1024][2048] (Vt, K-contiguous for P@V)
// WG = 256 threads = 8 waves; WG tile 32 rows x 256 cols; wave tile 16x64.
// ---------------------------------------------------------------------------
__global__ __launch_bounds__(256)
void gemm_qkv(const _Float16* __restrict__ xh, const _Float16* __restrict__ Wt,
              const float* __restrict__ bias, _Float16* __restrict__ out,
              int transposeOut) {
    const int lane = threadIdx.x & 31;
    const int wave = threadIdx.x >> 5;
    const int lm = lane & 15;
    const int lh = lane >> 4;
    const int waveRow = wave >> 2;                 // 0..1
    const int waveCol = wave & 3;                  // 0..3
    const int rb = blockIdx.x & 255;               // 256 row blocks of 32
    const int cb = blockIdx.x >> 8;                // 4 col blocks of 256
    const int row0 = rb * 32 + waveRow * 16;
    const int col0 = cb * 256 + waveCol * 64;

    const _Float16* aPtr = xh + (size_t)(row0 + lm) * DIMD + lh * 16;

    v8f acc[4] = {};
    float bcol[4];
#pragma unroll
    for (int nt = 0; nt < 4; ++nt) bcol[nt] = bias[col0 + nt * 16 + lm];

    for (int kk = 0; kk < DIMD; kk += 32) {
        AFrag a;
        a.h[0] = *(const v8h*)(aPtr + kk);         // K = kk + lh*16 .. +7
        a.h[1] = *(const v8h*)(aPtr + kk + 8);     // K = kk+8 + lh*16 .. +7
#pragma unroll
        for (int nt = 0; nt < 4; ++nt) {
            int n = col0 + nt * 16 + lm;
            v16h b = *(const v16h*)(Wt + (size_t)n * DIMD + kk + lh * 16);
            acc[nt] = __builtin_amdgcn_wmma_f32_16x16x32_f16(
                false, a.v, false, b, (short)0, acc[nt], false, false);
        }
    }

#pragma unroll
    for (int nt = 0; nt < 4; ++nt) {
        int n = col0 + nt * 16 + lm;
#pragma unroll
        for (int i = 0; i < 8; ++i) {
            int gr = row0 + i + lh * 8;            // global row (C/D layout: M = i + 8*half)
            float val = acc[nt][i] + bcol[nt];
            size_t addr;
            if (transposeOut) {
                int bb = gr >> 11, s = gr & (SEQ - 1);
                addr = (size_t)bb * DIMD * SEQ + (size_t)n * SEQ + s;
            } else {
                addr = (size_t)gr * DIMD + n;
            }
            out[addr] = (_Float16)val;
        }
    }
}

// ---------------------------------------------------------------------------
// Attention: one WG handles 16 query rows of one batch.
//  Phase 1: scores[16][2048] = (Q_blk @ K^T)*scale + mask  -> LDS (f32)
//  Phase 2: row softmax (max, exp in place, rowsum) cooperatively
//  Phase 3: out = (P @ V) / rowsum  (P converted f32->f16 from LDS)
// ---------------------------------------------------------------------------
__global__ __launch_bounds__(256)
void attn_kernel(const _Float16* __restrict__ Q, const _Float16* __restrict__ K,
                 const _Float16* __restrict__ Vt, const float* __restrict__ mask,
                 float* __restrict__ out) {
    extern __shared__ float lds[];
    float* sc     = lds;                           // [16][LSTR]
    float* red    = lds + 16 * LSTR;               // [16][16] reduction scratch
    float* rowsum = red + 256;                     // [16]

    const int tid  = threadIdx.x;
    const int lane = tid & 31;
    const int wave = tid >> 5;
    const int lm = lane & 15;
    const int lh = lane >> 4;

    const int row0 = blockIdx.x * 16;              // global query row base
    const int b    = row0 >> 11;                   // batch
    const int qs   = row0 & (SEQ - 1);             // row within batch
    const _Float16* Kb = K  + (size_t)b * SEQ * DIMD;
    const _Float16* Vb = Vt + (size_t)b * DIMD * SEQ;

    // ---- phase 1: scores ----
    const _Float16* qPtr = Q + (size_t)(row0 + lm) * DIMD + lh * 16;
    for (int g = 0; g < 2; ++g) {                  // 16 kv-tiles per wave, 2 groups of 8
        v8f acc[8] = {};
        int tile0 = wave * 16 + g * 8;
        for (int kk = 0; kk < DIMD; kk += 32) {
            AFrag a;
            a.h[0] = *(const v8h*)(qPtr + kk);
            a.h[1] = *(const v8h*)(qPtr + kk + 8);
#pragma unroll
            for (int t = 0; t < 8; ++t) {
                int n = (tile0 + t) * 16 + lm;     // kv index (B column)
                v16h bf = *(const v16h*)(Kb + (size_t)n * DIMD + kk + lh * 16);
                acc[t] = __builtin_amdgcn_wmma_f32_16x16x32_f16(
                    false, a.v, false, bf, (short)0, acc[t], false, false);
            }
        }
#pragma unroll
        for (int t = 0; t < 8; ++t) {
            int n = (tile0 + t) * 16 + lm;
#pragma unroll
            for (int i = 0; i < 8; ++i) {
                int m = i + lh * 8;
                float v = acc[t][i] * ATT_SCALE + mask[(size_t)(qs + m) * SEQ + n];
                sc[m * LSTR + n] = v;
            }
        }
    }
    __syncthreads();

    // ---- phase 2: softmax ----
    {
        int r = tid >> 4;                          // row 0..15
        int c = tid & 15;                          // 16 threads per row
        float mx = -3.4e38f;
        for (int j = c; j < SEQ; j += 16) mx = fmaxf(mx, sc[r * LSTR + j]);
        red[r * 16 + c] = mx;
        __syncthreads();
        mx = red[r * 16];
#pragma unroll
        for (int j = 1; j < 16; ++j) mx = fmaxf(mx, red[r * 16 + j]);
        float sum = 0.f;
        for (int j = c; j < SEQ; j += 16) {
            float e = __expf(sc[r * LSTR + j] - mx);
            sc[r * LSTR + j] = e;
            sum += e;
        }
        __syncthreads();                           // all max reads of `red` done
        red[r * 16 + c] = sum;
        __syncthreads();
        if (c == 0) {
            float s = 0.f;
#pragma unroll
            for (int j = 0; j < 16; ++j) s += red[r * 16 + j];
            rowsum[r] = s;
        }
        __syncthreads();
    }

    // ---- phase 3: out = (P @ V) / rowsum ----
    {
        v8f acc[8] = {};
        int dt0 = wave * 8;                        // 64 d-tiles, 8 per wave
        const float* pPtr = sc + (size_t)lm * LSTR + lh * 16;
        for (int kv = 0; kv < SEQ; kv += 32) {
            AFrag a;
            v8f p0 = *(const v8f*)(pPtr + kv);
            v8f p1 = *(const v8f*)(pPtr + kv + 8);
            a.h[0] = __builtin_convertvector(p0, v8h);
            a.h[1] = __builtin_convertvector(p1, v8h);
#pragma unroll
            for (int t = 0; t < 8; ++t) {
                int d = (dt0 + t) * 16 + lm;
                v16h bf = *(const v16h*)(Vb + (size_t)d * SEQ + kv + lh * 16);
                acc[t] = __builtin_amdgcn_wmma_f32_16x16x32_f16(
                    false, a.v, false, bf, (short)0, acc[t], false, false);
            }
        }
#pragma unroll
        for (int t = 0; t < 8; ++t) {
            int d = (dt0 + t) * 16 + lm;
#pragma unroll
            for (int i = 0; i < 8; ++i) {
                int m = i + lh * 8;
                out[(size_t)(row0 + m) * DIMD + d] = acc[t][i] / rowsum[m];
            }
        }
    }
}

// ---------------------------------------------------------------------------
extern "C" void kernel_launch(void* const* d_in, const int* in_sizes, int n_in,
                              void* d_out, int out_size, void* d_ws, size_t ws_size,
                              hipStream_t stream) {
    const float* x    = (const float*)d_in[0];
    const float* mask = (const float*)d_in[1];
    const float* Wq   = (const float*)d_in[2];
    const float* bq   = (const float*)d_in[3];
    const float* Wk   = (const float*)d_in[4];
    const float* bk   = (const float*)d_in[5];
    const float* Wv   = (const float*)d_in[6];
    const float* bv   = (const float*)d_in[7];
    float* out = (float*)d_out;

    // workspace carve-up (all f16): x 16MB, 3 x Wt 2MB, Q 16MB, K 16MB, Vt 16MB = 70MB
    char* ws = (char*)d_ws;
    _Float16* xh  = (_Float16*)ws;  ws += (size_t)NROW * DIMD * 2;
    _Float16* Wtq = (_Float16*)ws;  ws += (size_t)DIMD * DIMD * 2;
    _Float16* Wtk = (_Float16*)ws;  ws += (size_t)DIMD * DIMD * 2;
    _Float16* Wtv = (_Float16*)ws;  ws += (size_t)DIMD * DIMD * 2;
    _Float16* Qh  = (_Float16*)ws;  ws += (size_t)NROW * DIMD * 2;
    _Float16* Kh  = (_Float16*)ws;  ws += (size_t)NROW * DIMD * 2;
    _Float16* Vth = (_Float16*)ws;  ws += (size_t)NROW * DIMD * 2;

    // 1) conversions
    cvt_f32_f16<<<(NROW * DIMD) / 1024, 256, 0, stream>>>(x, xh, NROW * DIMD);
    cvt_w_transpose<<<(DIMD * DIMD) / 256, 256, 0, stream>>>(Wq, Wtq);
    cvt_w_transpose<<<(DIMD * DIMD) / 256, 256, 0, stream>>>(Wk, Wtk);
    cvt_w_transpose<<<(DIMD * DIMD) / 256, 256, 0, stream>>>(Wv, Wtv);

    // 2) Q/K/V projections (grid: 256 row-blocks x 4 col-blocks)
    gemm_qkv<<<1024, 256, 0, stream>>>(xh, Wtq, bq, Qh, 0);
    gemm_qkv<<<1024, 256, 0, stream>>>(xh, Wtk, bk, Kh, 0);
    gemm_qkv<<<1024, 256, 0, stream>>>(xh, Wtv, bv, Vth, 1);  // V stored [b][d][s]

    // 3) attention: one WG per 16 query rows
    size_t smem = (size_t)(16 * LSTR + 256 + 16) * sizeof(float);  // ~132.7 KB
    attn_kernel<<<NROW / 16, 256, smem, stream>>>(Qh, Kh, Vth, mask, out);
}
